// NeuralInelasticModel_55155970015481
// MI455X (gfx1250) — matrix operand order repaired and checked by reading, not compiled, not run on hardware
//
#include <hip/hip_runtime.h>

// MI455X / gfx1250, wave32. All heavy math routed through v_wmma_f32_16x16x32_bf16.
// v2: A-operand tiles hoisted into registers (loop-invariant across nt/i),
//     nt-loops unrolled x2 for two independent WMMA accumulation chains.

typedef __attribute__((ext_vector_type(16))) __bf16 v16bf;
typedef __attribute__((ext_vector_type(8)))  float  v8f;

#define S_TOTAL 65536   // 64 * 1024 samples
#define H 256
#define NS 6
#define NI 8

__device__ __forceinline__ unsigned short f2bf(float f) {
  unsigned int u = __float_as_uint(f);
  u += 0x7FFFu + ((u >> 16) & 1u);          // round-to-nearest-even
  return (unsigned short)(u >> 16);
}
__device__ __forceinline__ float bf2f(unsigned short h) {
  return __uint_as_float(((unsigned int)h) << 16);
}

union TileU { unsigned int u[8]; v16bf v; };

// B-operand tile (K x N = 32 x 16 bf16). Contiguous-K layout used here: lanes 0-15 hold N=lane,
// K=0..15 packed 2-per-VGPR; lanes 16-31 hold K=16..31. Caller passes a
// per-lane pointer to 16 contiguous bf16 (32B, aligned) -> 2x global_load_b128.
__device__ __forceinline__ v16bf ld_b_tile(const unsigned short* p) {
  uint4 a = *(const uint4*)p;
  uint4 b = *(const uint4*)(p + 8);
  TileU t;
  t.u[0] = a.x; t.u[1] = a.y; t.u[2] = a.z; t.u[3] = a.w;
  t.u[4] = b.x; t.u[5] = b.y; t.u[6] = b.z; t.u[7] = b.w;
  return t.v;
}

// A-operand tile (M x K = 16 x 32 bf16) from row-major [16][256] LDS buffer.
// Layout: lane L<16: row M=L, K 0..15 contiguous;
// lane L>=16: row M=L-16, K 16..31 (matches B tile). Two 16B ds_load_b128 per lane.
__device__ __forceinline__ v16bf ld_a_tile(const unsigned short* buf, int kt, int lane) {
  int row = lane & 15, hi = lane >> 4;
  const unsigned short* p0 = buf + row * H + kt * 32 + hi * 16;
  uint4 a = *(const uint4*)p0;
  uint4 b = *(const uint4*)(p0 + 8);
  TileU t;
  t.u[0] = a.x; t.u[1] = a.y; t.u[2] = a.z; t.u[3] = a.w;
  t.u[4] = b.x; t.u[5] = b.y; t.u[6] = b.z; t.u[7] = b.w;
  return t.v;
}

// Precompute bf16 operand matrices (all L2-resident afterwards, < 1 MB total):
//   w2bf[n][k]     = bf16(w2[n][k])                        (layer-2 B tiles)
//   Rt[i][h1][h2]  = bf16(w3[i][h2] * w2[h2][h1])          (Jacobian B tiles)
//   w1t[j][h]      = bf16(w1[h][j]) for j<8, 0 for j=8..15 (final contraction B)
__global__ void prep_kernel(const float* __restrict__ w1, const float* __restrict__ w2,
                            const float* __restrict__ w3,
                            unsigned short* __restrict__ w2bf,
                            unsigned short* __restrict__ Rt,
                            unsigned short* __restrict__ w1t) {
  int idx = blockIdx.x * blockDim.x + threadIdx.x;   // 0 .. 6*256*256-1
  int i   = idx >> 16;
  int rem = idx & 65535;
  int h1  = rem >> 8, h2 = rem & 255;
  Rt[idx] = f2bf(w3[i * H + h2] * w2[h2 * H + h1]);
  if (idx < H * H) w2bf[idx] = f2bf(w2[idx]);
  if (idx < 16 * H) {
    int j = idx >> 8, h = idx & 255;
    w1t[idx] = (j < NI) ? f2bf(w1[h * NI + j]) : (unsigned short)0;
  }
}

__launch_bounds__(128)
__global__ void fused_kernel(const float* __restrict__ y,  const float* __restrict__ er,
                             const float* __restrict__ Tm, const float* __restrict__ w1,
                             const float* __restrict__ w3, const float* __restrict__ b1,
                             const float* __restrict__ b2, const float* __restrict__ b3,
                             const unsigned short* __restrict__ w2bf,
                             const unsigned short* __restrict__ Rt,
                             const unsigned short* __restrict__ w1t,
                             float* __restrict__ out_ydot, float* __restrict__ out_dy,
                             float* __restrict__ out_de,   float* __restrict__ out_dT) {
  extern __shared__ char smem[];
  const int wave = threadIdx.x >> 5;
  const int lane = threadIdx.x & 31;
  const int col  = lane & 15;
  const int hi   = lane >> 4;

  // per-wave LDS region: 4 x (16x256 bf16) buffers + x staging (33,280 B)
  unsigned short* v1b  = (unsigned short*)(smem + wave * 33280);
  unsigned short* m1b  = v1b + 4096;
  unsigned short* m2b  = m1b + 4096;
  unsigned short* auxb = m2b + 4096;       // v2 (forward), then masked-C scratch
  float* xls = (float*)(auxb + 4096);

  const int sbase = blockIdx.x * 64 + wave * 16;

  // ---- phase 0: gather x = [y, erate, T] for 16 samples ----
  for (int f = lane; f < 16 * NI; f += 32) {
    int s = f >> 3, c = f & 7, g = sbase + s;
    xls[f] = (c < NS) ? y[g * NS + c] : ((c == NS) ? er[g] : Tm[g]);
  }
  __syncthreads();

  // ---- phase 1: layer 1 (8-wide: fp32 VALU), write v1/m1 as bf16 ----
  for (int q = 0; q < 8; ++q) {
    int h = lane + q * 32;
    float wr[NI];
    #pragma unroll
    for (int j = 0; j < NI; ++j) wr[j] = w1[h * NI + j];
    float bb = b1[h];
    for (int s = 0; s < 16; ++s) {
      float z = bb;
      #pragma unroll
      for (int j = 0; j < NI; ++j) z = fmaf(wr[j], xls[s * NI + j], z);
      v1b[s * H + h] = f2bf(z > 0.f ? z : 0.f);
      m1b[s * H + h] = (z > 0.f) ? (unsigned short)0x3F80 : (unsigned short)0;
    }
  }
  __syncthreads();

  // ---- phase 2: layer 2  z2 = v1 @ w2^T + b2  (bf16 WMMA, f32 acc) ----
  {
    v16bf Av[8];                                 // hoist A (v1) tiles: reused by all 16 nt
    #pragma unroll
    for (int kt = 0; kt < 8; ++kt) Av[kt] = ld_a_tile(v1b, kt, lane);

    #pragma unroll 2
    for (int nt = 0; nt < 16; ++nt) {
      int n = nt * 16 + col;
      float bias = b2[n];
      v8f acc;
      #pragma unroll
      for (int r = 0; r < 8; ++r) acc[r] = bias;
      #pragma unroll
      for (int kt = 0; kt < 8; ++kt) {
        v16bf B = ld_b_tile(w2bf + n * H + kt * 32 + hi * 16);
        acc = __builtin_amdgcn_wmma_f32_16x16x32_bf16(false, Av[kt], false, B, (short)0, acc, false, false);
      }
      #pragma unroll
      for (int r = 0; r < 8; ++r) {            // D layout: lane holds (s = r+hi*8, n)
        float z = acc[r];
        int s = r + hi * 8;
        m2b[s * H + n]  = (z > 0.f) ? (unsigned short)0x3F80 : (unsigned short)0;
        auxb[s * H + n] = f2bf(z > 0.f ? z : 0.f);   // v2
      }
    }
  }
  __syncthreads();

  // ---- phase 3: ydot = w3 @ v2 + b3 (6-wide: fp32 VALU) ----
  for (int q = 0; q < 3; ++q) {
    int p = lane + q * 32;                   // 96 (s,i) pairs over 32 lanes
    int s = p / NS, i = p % NS;
    float acc = b3[i];
    #pragma unroll 4
    for (int h = 0; h < H; ++h)
      acc = fmaf(bf2f(auxb[s * H + h]), w3[i * H + h], acc);
    out_ydot[(sbase + s) * NS + i] = acc;
  }
  __syncthreads();

  // ---- phase 4: Jacobian. For each output row i:
  //   C = m2 @ Rt_i^T  (16x256, WMMA), mask by m1, then J_i = Cm @ w1t^T ----
  v16bf Am[8];                                 // hoist A (m2) tiles: reused by all 6x16 (i,nt)
  #pragma unroll
  for (int kt = 0; kt < 8; ++kt) Am[kt] = ld_a_tile(m2b, kt, lane);

  for (int i = 0; i < NS; ++i) {
    const unsigned short* Rti = Rt + i * (H * H);
    #pragma unroll 2
    for (int nt = 0; nt < 16; ++nt) {
      int n = nt * 16 + col;
      v8f acc = {0.f, 0.f, 0.f, 0.f, 0.f, 0.f, 0.f, 0.f};
      #pragma unroll
      for (int kt = 0; kt < 8; ++kt) {
        v16bf B = ld_b_tile(Rti + n * H + kt * 32 + hi * 16);
        acc = __builtin_amdgcn_wmma_f32_16x16x32_bf16(false, Am[kt], false, B, (short)0, acc, false, false);
      }
      #pragma unroll
      for (int r = 0; r < 8; ++r) {
        int s = r + hi * 8;
        auxb[s * H + n] = f2bf(m1b[s * H + n] ? acc[r] : 0.f);   // apply diag(m1)
      }
    }
    __syncthreads();
    v8f jacc = {0.f, 0.f, 0.f, 0.f, 0.f, 0.f, 0.f, 0.f};
    #pragma unroll
    for (int kt = 0; kt < 8; ++kt) {
      v16bf A = ld_a_tile(auxb, kt, lane);     // per-i layout transform, must reload
      v16bf B = ld_b_tile(w1t + col * H + kt * 32 + hi * 16);    // cols 8..15 are zero
      jacc = __builtin_amdgcn_wmma_f32_16x16x32_bf16(false, A, false, B, (short)0, jacc, false, false);
    }
    int j = col;
    #pragma unroll
    for (int r = 0; r < 8; ++r) {
      int g = sbase + r + hi * 8;
      float v = jacc[r];
      if (j < NS)            out_dy[g * (NS * NS) + i * NS + j] = v;
      else if (j == NS)      out_de[g * NS + i] = v;
      else if (j == NS + 1)  out_dT[g * NS + i] = v;
    }
    __syncthreads();
  }
}

extern "C" void kernel_launch(void* const* d_in, const int* in_sizes, int n_in,
                              void* d_out, int out_size, void* d_ws, size_t ws_size,
                              hipStream_t stream) {
  (void)in_sizes; (void)n_in; (void)out_size; (void)ws_size;
  // input order: t(unused), y, erate, T, w1, w2, w3, b1, b2, b3
  const float* y  = (const float*)d_in[1];
  const float* er = (const float*)d_in[2];
  const float* Tm = (const float*)d_in[3];
  const float* w1 = (const float*)d_in[4];
  const float* w2 = (const float*)d_in[5];
  const float* w3 = (const float*)d_in[6];
  const float* b1 = (const float*)d_in[7];
  const float* b2 = (const float*)d_in[8];
  const float* b3 = (const float*)d_in[9];

  unsigned short* w2bf = (unsigned short*)d_ws;        // 128 KB
  unsigned short* Rt   = w2bf + H * H;                 // 768 KB
  unsigned short* w1t  = Rt + NS * H * H;              // 8 KB   (total < 1 MB ws)

  float* out      = (float*)d_out;                     // concatenated outputs
  float* out_ydot = out;
  float* out_dy   = out + S_TOTAL * NS;
  float* out_de   = out_dy + S_TOTAL * NS * NS;
  float* out_dT   = out_de + S_TOTAL * NS;

  prep_kernel<<<(NS * H * H) / 256, 256, 0, stream>>>(w1, w2, w3, w2bf, Rt, w1t);
  // 1024 blocks x 128 threads (4 waves x 16 samples); 133,120 B dynamic LDS (< 320 KB WGP)
  fused_kernel<<<S_TOTAL / 64, 128, 4 * 33280, stream>>>(
      y, er, Tm, w1, w3, b1, b2, b3, w2bf, Rt, w1t,
      out_ydot, out_dy, out_de, out_dT);
}
